// PerformerAttention_75368086110617
// MI455X (gfx1250) — compile-verified, matrix operand order checked by reading, not run on hardware
//
#include <hip/hip_runtime.h>
#include <hip/hip_bf16.h>
#include <cstdint>

// Performer attention for MI455X (gfx1250, wave32, WMMA bf16 16x16x32).
// - All five GEMMs go through v_wmma_f32_16x16x32_bf16, fp32 accumulation.
// - LDS tiles stored in fragment order (A: [m][k], B transposed: [n][k]) so
//   each WMMA fragment is exactly two ds_load_b128 per lane.
// - Double-buffered LDS, one barrier per K-step, hoisted address arithmetic.
// - bf16 conversion via hardware cvt (static_cast<__bf16>, RNE).
// - Async global->LDS staging (GLOBAL_LOAD_ASYNC_TO_LDS_B128 / ASYNCcnt) where
//   the copy is layout-preserving bf16; global_prefetch for fp32 staging.
// - Split-K (deterministic fp32 partials, no atomics) for the small kptv GEMM.

#define DEVFN __device__ __forceinline__

typedef __attribute__((ext_vector_type(16))) __bf16 v16bf;
typedef __attribute__((ext_vector_type(8)))  __bf16 v8bf;
typedef __attribute__((ext_vector_type(8)))  float  v8f;
typedef int v4i __attribute__((vector_size(16)));   // matches builtin proto

constexpr int Bn = 4, Tn = 8192, Dn = 512, Mn = 256;
constexpr int KSPLIT_KPTV = 8;

#if defined(__has_builtin)
#if __has_builtin(__builtin_amdgcn_global_load_async_to_lds_b128)
#define HAVE_ASYNC_LDS 1
#endif
#endif
#ifndef HAVE_ASYNC_LDS
#define HAVE_ASYNC_LDS 0
#endif

DEVFN unsigned short f32_to_bf16_bits(float f) {
  return __builtin_bit_cast(unsigned short, static_cast<__bf16>(f)); // HW cvt, RNE
}
DEVFN float bf16_bits_to_f32(unsigned short h) {
  return __builtin_bit_cast(float, (uint32_t)h << 16);
}

// 8-element (16-byte) chunk loads; fp32 sources converted with HW cvt.
DEVFN v8bf load8(const unsigned short* p) {
  return *reinterpret_cast<const v8bf*>(p);        // global_load_b128
}
DEVFN v8bf load8(const float* p) {
  v8bf r;
  #pragma unroll
  for (int e = 0; e < 8; ++e) r[e] = static_cast<__bf16>(p[e]);
  return r;
}

#if HAVE_ASYNC_LDS
DEVFN void async_cp16(const unsigned short* g, unsigned short* l) {
  __builtin_amdgcn_global_load_async_to_lds_b128(
      (__attribute__((address_space(1))) v4i*)g,
      (__attribute__((address_space(3))) v4i*)l, 0, 0);
}
#endif
DEVFN void wait_async0() {
#if HAVE_ASYNC_LDS
#if __has_builtin(__builtin_amdgcn_s_wait_asynccnt)
  __builtin_amdgcn_s_wait_asynccnt(0);
#else
  asm volatile("s_wait_asynccnt 0" ::: "memory");
#endif
#endif
}

enum { EP_BIAS_BF16 = 0, EP_FEAT = 1, EP_DIV_BF16 = 3, EP_BIAS_F32 = 4, EP_PLAIN_F32 = 5 };

// C[M,N]: A is [Mrows,K] (ATRANS=0) or [K,Mrows] (ATRANS=1). B is [K,N]
// (BTRANS=0) or [N,K] (BTRANS=1, C = A*B^T).
// aux: bias[N] (EP_BIAS_*) or rowv[Mrows] (EP_FEAT/EP_DIV).
// blockIdx.z = zb*ksplit + ks;  A/B/aux stride by zb, C strides by raw z.
template<int MODE, bool ATRANS, bool BTRANS, typename TA, typename TB, typename TC>
__global__ __launch_bounds__(256)
void wmma_gemm(const TA* __restrict__ Ag, const TB* __restrict__ Bg,
               TC* __restrict__ Cg, const float* __restrict__ aux,
               int lda, int ldb, int ldc, int K, int ksplit,
               size_t strideA, size_t strideB, size_t strideC, size_t strideAux,
               float scale)
{
  constexpr int BM = 128, BN = 128, BK = 32, LDT = BK + 8; // 80B rows, 16B aligned
  __shared__ __align__(16) unsigned short As[2][BM][LDT];
  __shared__ __align__(16) unsigned short Bs[2][BN][LDT];  // B stored transposed: [n][k]

  constexpr bool asyncA = HAVE_ASYNC_LDS && !ATRANS && __is_same(TA, unsigned short);
  constexpr bool asyncB = HAVE_ASYNC_LDS &&  BTRANS && __is_same(TB, unsigned short);

  const int tid  = threadIdx.x;
  const int lane = tid & 31;
  const int wave = tid >> 5;            // 8 waves: 4 (M) x 2 (N)
  const int wm   = wave >> 1;
  const int wn   = wave & 1;
  const int lr   = lane & 15;           // row/col within a 16x16 tile
  const int kb   = (lane >> 4) * 8;     // K-half per CDNA5 16-bit layout
  const int hi   = lane >> 4;

  const int n0 = blockIdx.x * BN;
  const int m0 = blockIdx.y * BM;
  const int z  = blockIdx.z;
  const int zb = z / ksplit;
  const int ks = z - zb * ksplit;
  const int kChunk = K / ksplit;
  const int kStart = ks * kChunk;
  const int kEnd   = kStart + kChunk;

  const TA* A = Ag + (size_t)zb * strideA;
  const TB* B = Bg + (size_t)zb * strideB;
  TC*       C = Cg + (size_t)z  * strideC;
  const float* auxp = aux ? (aux + (size_t)zb * strideAux) : nullptr;

  // ---- per-thread staging pointers, hoisted out of the K loop ----
  const TA* aSrc[2]; int aDstOff[2];
  const TB* bSrc[2]; int bDstOff[2];
  #pragma unroll
  for (int it = 0; it < 2; ++it) {
    int c = tid + it * 256;                       // 512 chunks of 8 elements
    if (!ATRANS) {                                // A[m][k]: contiguous along K
      int m = c >> 2, kc = (c & 3) * 8;
      aSrc[it]   = A + (size_t)(m0 + m) * lda + (kStart + kc);
      aDstOff[it] = m * LDT + kc;
    } else {                                      // A[k][m]: contiguous along M
      int kk = c >> 4, mb = (c & 15) * 8;
      aSrc[it]   = A + (size_t)(kStart + kk) * lda + (m0 + mb);
      aDstOff[it] = mb * LDT + kk;
    }
    if (BTRANS) {                                 // B[n][k]: contiguous along K
      int n = c >> 2, kc = (c & 3) * 8;
      bSrc[it]   = B + (size_t)(n0 + n) * ldb + (kStart + kc);
      bDstOff[it] = n * LDT + kc;
    } else {                                      // B[k][n]: contiguous along N
      int kk = c >> 4, nb = (c & 15) * 8;
      bSrc[it]   = B + (size_t)(kStart + kk) * ldb + (n0 + nb);
      bDstOff[it] = nb * LDT + kk;
    }
  }

  auto stageA = [&](int buf) {
    unsigned short* base = &As[buf][0][0];
    #pragma unroll
    for (int it = 0; it < 2; ++it) {
      if constexpr (!ATRANS) {
        if constexpr (asyncA) {
          async_cp16(aSrc[it], base + aDstOff[it]);
        } else {
          if constexpr (__is_same(TA, float))
            __builtin_prefetch(aSrc[it] + 2 * BK, 0, 1);
          *(v8bf*)(base + aDstOff[it]) = load8(aSrc[it]);
        }
        aSrc[it] += BK;
      } else {
        v8bf v = load8(aSrc[it]);
        #pragma unroll
        for (int e = 0; e < 8; ++e)
          base[aDstOff[it] + e * LDT] = __builtin_bit_cast(unsigned short, v[e]);
        aSrc[it] += (size_t)BK * lda;
      }
    }
  };
  auto stageB = [&](int buf) {
    unsigned short* base = &Bs[buf][0][0];
    #pragma unroll
    for (int it = 0; it < 2; ++it) {
      if constexpr (BTRANS) {
        if constexpr (asyncB) {
          async_cp16(bSrc[it], base + bDstOff[it]);
        } else {
          if constexpr (__is_same(TB, float))
            __builtin_prefetch(bSrc[it] + 2 * BK, 0, 1);
          *(v8bf*)(base + bDstOff[it]) = load8(bSrc[it]);
        }
        bSrc[it] += BK;
      } else {
        v8bf v = load8(bSrc[it]);
        #pragma unroll
        for (int e = 0; e < 8; ++e)
          base[bDstOff[it] + e * LDT] = __builtin_bit_cast(unsigned short, v[e]);
        bSrc[it] += (size_t)BK * ldb;
      }
    }
  };

  v8f acc[2][4];
  #pragma unroll
  for (int mi = 0; mi < 2; ++mi)
    #pragma unroll
    for (int ni = 0; ni < 4; ++ni)
      #pragma unroll
      for (int r = 0; r < 8; ++r) acc[mi][ni][r] = 0.0f;

  stageA(0);
  stageB(0);
  if constexpr (asyncA || asyncB) wait_async0();
  __syncthreads();

  int cur = 0;
  for (int k0 = kStart; k0 < kEnd; k0 += BK) {
    int nxt = cur ^ 1;
    if (k0 + BK < kEnd) { stageA(nxt); stageB(nxt); }

    // Fragments: two contiguous 16B runs per lane -> 2x ds_load_b128 each.
    union V16U { v16bf v; v8bf h[2]; };
    v16bf afr[2], bfr[4];
    #pragma unroll
    for (int mi = 0; mi < 2; ++mi) {
      int row = wm * 32 + mi * 16 + lr;
      V16U u;
      u.h[0] = *(const v8bf*)&As[cur][row][kb];
      u.h[1] = *(const v8bf*)&As[cur][row][kb + 16];
      afr[mi] = u.v;
    }
    #pragma unroll
    for (int ni = 0; ni < 4; ++ni) {
      int col = wn * 64 + ni * 16 + lr;
      V16U u;
      u.h[0] = *(const v8bf*)&Bs[cur][col][kb];
      u.h[1] = *(const v8bf*)&Bs[cur][col][kb + 16];
      bfr[ni] = u.v;
    }

    #pragma unroll
    for (int mi = 0; mi < 2; ++mi)
      #pragma unroll
      for (int ni = 0; ni < 4; ++ni)
        acc[mi][ni] = __builtin_amdgcn_wmma_f32_16x16x32_bf16(
            false, afr[mi], false, bfr[ni], (short)0, acc[mi][ni], false, false);

    if constexpr (asyncA || asyncB) wait_async0();
    __syncthreads();
    cur = nxt;
  }

  // Epilogue. C/D layout: n = l&15; m = (l>>4)*8 + r.
  #pragma unroll
  for (int mi = 0; mi < 2; ++mi) {
    #pragma unroll
    for (int ni = 0; ni < 4; ++ni) {
      int n = n0 + wn * 64 + ni * 16 + lr;
      float bn = 0.0f;
      if constexpr (MODE == EP_BIAS_BF16 || MODE == EP_BIAS_F32) bn = auxp[n];
      #pragma unroll
      for (int r = 0; r < 8; ++r) {
        int m = m0 + wm * 32 + mi * 16 + hi * 8 + r;
        float v = acc[mi][ni][r];
        size_t off = (size_t)m * ldc + n;
        if constexpr (MODE == EP_BIAS_BF16) {
          C[off] = f32_to_bf16_bits(v + bn);
        } else if constexpr (MODE == EP_FEAT) {
          C[off] = f32_to_bf16_bits(__expf(v - auxp[m]) * scale);
        } else if constexpr (MODE == EP_DIV_BF16) {
          C[off] = f32_to_bf16_bits(v / (auxp[m] + 1e-8f));
        } else if constexpr (MODE == EP_BIAS_F32) {
          C[off] = v + bn;
        } else { // EP_PLAIN_F32 (split-K partials)
          C[off] = v;
        }
      }
    }
  }
}

// 0.5 * sum(x^2) per row; one wave32 per row.
__global__ __launch_bounds__(256)
void rownorm_bf16(const unsigned short* __restrict__ X, int ld, int cols,
                  float* __restrict__ out, int rows)
{
  int row  = blockIdx.x * 8 + (threadIdx.x >> 5);
  int lane = threadIdx.x & 31;
  if (row >= rows) return;
  const unsigned short* p = X + (size_t)row * ld;
  float s = 0.0f;
  for (int c = lane; c < cols; c += 32) {
    float v = bf16_bits_to_f32(p[c]);
    s += v * v;
  }
  #pragma unroll
  for (int off = 16; off > 0; off >>= 1) s += __shfl_xor(s, off, 32);
  if (lane == 0) out[row] = 0.5f * s;
}

// ksum[b][m] = sum_t kp[b][t][m]; one thread per (b,m), coalesced over m.
__global__ __launch_bounds__(256)
void colsum_bf16(const unsigned short* __restrict__ KP, float* __restrict__ out)
{
  int b = blockIdx.x;
  int m = threadIdx.x;                  // blockDim == Mn == 256
  const unsigned short* p = KP + (size_t)b * Tn * Mn + m;
  float s = 0.0f;
  for (int t = 0; t < Tn; ++t) { s += bf16_bits_to_f32(*p); p += Mn; }
  out[b * Mn + m] = s;
}

// denom[row] = qp[row] . ksum[batch]; one wave32 per row.
__global__ __launch_bounds__(256)
void denom_kernel(const unsigned short* __restrict__ QP,
                  const float* __restrict__ ksum, float* __restrict__ out)
{
  int row  = blockIdx.x * 8 + (threadIdx.x >> 5);
  int lane = threadIdx.x & 31;
  int b = row >> 13;                    // row / Tn
  const unsigned short* p = QP + (size_t)row * Mn;
  const float* ks = ksum + b * Mn;
  float s = 0.0f;
  #pragma unroll
  for (int i = 0; i < Mn / 32; ++i) {
    int c = lane + i * 32;
    s += bf16_bits_to_f32(p[c]) * ks[c];
  }
  #pragma unroll
  for (int off = 16; off > 0; off >>= 1) s += __shfl_xor(s, off, 32);
  if (lane == 0) out[row] = s;
}

// kptv[b][d][m] = sum_ks part[b*KSPLIT+ks][d][m]  (deterministic reduce + cvt)
__global__ __launch_bounds__(256)
void reduce_kptv(const float* __restrict__ part, unsigned short* __restrict__ out)
{
  size_t i = (size_t)blockIdx.x * 256 + threadIdx.x;   // Bn*Dn*Mn total
  int b = (int)(i / (Dn * Mn));
  size_t r = i - (size_t)b * (Dn * Mn);
  float s = 0.0f;
  #pragma unroll
  for (int ks = 0; ks < KSPLIT_KPTV; ++ks)
    s += part[((size_t)b * KSPLIT_KPTV + ks) * (Dn * Mn) + r];
  out[i] = f32_to_bf16_bits(s);
}

extern "C" void kernel_launch(void* const* d_in, const int* in_sizes, int n_in,
                              void* d_out, int out_size, void* d_ws, size_t ws_size,
                              hipStream_t stream) {
  (void)in_sizes; (void)n_in; (void)out_size; (void)ws_size;
  const float* x      = (const float*)d_in[0]; // [B,T,D]
  const float* w_qkv  = (const float*)d_in[1]; // [3D,D]
  const float* b_qkv  = (const float*)d_in[2]; // [3D]
  const float* w_proj = (const float*)d_in[3]; // [D,D]
  const float* b_proj = (const float*)d_in[4]; // [D]
  const float* w_feat = (const float*)d_in[5]; // [M,D]
  float* out = (float*)d_out;                  // [B,T,D] fp32

  char* wsp = (char*)d_ws;
  auto take = [&](size_t bytes) {
    char* p = wsp;
    wsp += (bytes + 255) & ~(size_t)255;
    return p;
  };
  unsigned short* qkvb  = (unsigned short*)take((size_t)Bn * Tn * 3 * Dn * 2); // k|q|v bf16
  unsigned short* kp    = (unsigned short*)take((size_t)Bn * Tn * Mn * 2);
  unsigned short* qp    = (unsigned short*)take((size_t)Bn * Tn * Mn * 2);
  unsigned short* kptvb = (unsigned short*)take((size_t)Bn * Dn * Mn * 2);
  unsigned short* ybuf  = (unsigned short*)take((size_t)Bn * Tn * Dn * 2);
  float* kptv_part = (float*)take((size_t)Bn * KSPLIT_KPTV * Dn * Mn * 4);
  float* xdk  = (float*)take((size_t)Bn * Tn * 4);
  float* xdq  = (float*)take((size_t)Bn * Tn * 4);
  float* ksum = (float*)take((size_t)Bn * Mn * 4);
  float* den  = (float*)take((size_t)Bn * Tn * 4);

  const int rowsBT = Bn * Tn;           // 32768

  // 1) qkv = x @ w_qkv^T + b_qkv  -> bf16 [B*T, 3D]  (split order: k,q,v)
  wmma_gemm<EP_BIAS_BF16, false, true, float, float, unsigned short>
      <<<dim3(3 * Dn / 128, rowsBT / 128, 1), 256, 0, stream>>>(
          x, w_qkv, qkvb, b_qkv, Dn, Dn, 3 * Dn, Dn, 1, 0, 0, 0, 0, 1.0f);

  // 2) xd = 0.5*|k|^2 and 0.5*|q|^2 per row
  rownorm_bf16<<<dim3(rowsBT / 8), 256, 0, stream>>>(qkvb + 0,  3 * Dn, Dn, xdk, rowsBT);
  rownorm_bf16<<<dim3(rowsBT / 8), 256, 0, stream>>>(qkvb + Dn, 3 * Dn, Dn, xdq, rowsBT);

  // 3) kp/qp = exp(k@w_feat^T - xd) / sqrt(M)  -> bf16 [B*T, M]
  wmma_gemm<EP_FEAT, false, true, unsigned short, float, unsigned short>
      <<<dim3(Mn / 128, rowsBT / 128, 1), 256, 0, stream>>>(
          qkvb + 0, w_feat, kp, xdk, 3 * Dn, Dn, Mn, Dn, 1, 0, 0, 0, 0, 0.0625f);
  wmma_gemm<EP_FEAT, false, true, unsigned short, float, unsigned short>
      <<<dim3(Mn / 128, rowsBT / 128, 1), 256, 0, stream>>>(
          qkvb + Dn, w_feat, qp, xdq, 3 * Dn, Dn, Mn, Dn, 1, 0, 0, 0, 0, 0.0625f);

  // 4) ksum[b,m] = sum_t kp[b,t,m]
  colsum_bf16<<<dim3(Bn), 256, 0, stream>>>(kp, ksum);

  // 5) denom[b,t] = qp[b,t,:] . ksum[b,:]
  denom_kernel<<<dim3(rowsBT / 8), 256, 0, stream>>>(qp, ksum, den);

  // 6) kptv partials: part[b,ks] = v_b[chunk]^T @ kp_b[chunk]  (split-K)
  wmma_gemm<EP_PLAIN_F32, true, false, unsigned short, unsigned short, float>
      <<<dim3(Mn / 128, Dn / 128, Bn * KSPLIT_KPTV), 256, 0, stream>>>(
          qkvb + 2 * Dn, kp, kptv_part, nullptr, 3 * Dn, Mn, Mn, Tn, KSPLIT_KPTV,
          (size_t)Tn * 3 * Dn, (size_t)Tn * Mn, (size_t)Dn * Mn, 0, 1.0f);
  reduce_kptv<<<dim3((Bn * Dn * Mn) / 256), 256, 0, stream>>>(kptv_part, kptvb);

  // 7) y[b] = (qp_b @ kptv_b^T) / (denom + 1e-8)  -> bf16 [T, D] per batch
  wmma_gemm<EP_DIV_BF16, false, true, unsigned short, unsigned short, unsigned short>
      <<<dim3(Dn / 128, Tn / 128, Bn), 256, 0, stream>>>(
          qp, kptvb, ybuf, den, Mn, Mn, Dn, Mn, 1,
          (size_t)Tn * Mn, (size_t)Dn * Mn, (size_t)Tn * Dn, (size_t)Tn, 1.0f);

  // 8) out = ybuf @ w_proj^T + b_proj  -> fp32 [B*T, D]
  wmma_gemm<EP_BIAS_F32, false, true, unsigned short, float, float>
      <<<dim3(Dn / 128, rowsBT / 128, 1), 256, 0, stream>>>(
          ybuf, w_proj, out, b_proj, Dn, Dn, Dn, Dn, 1, 0, 0, 0, 0, 1.0f);
}
